// OmniMoERouter_75514114998538
// MI455X (gfx1250) — matrix operand order repaired and checked by reading, not compiled, not run on hardware
//
#include <hip/hip_runtime.h>

typedef __attribute__((ext_vector_type(16))) __bf16       v16bf;
typedef __attribute__((ext_vector_type(2)))  __bf16       v2bf;
typedef __attribute__((ext_vector_type(8)))  float        v8f;
typedef __attribute__((ext_vector_type(8)))  float        v8ff;
typedef __attribute__((ext_vector_type(16))) float        v16f;
typedef __attribute__((ext_vector_type(2)))  float        v2ff;
typedef __attribute__((ext_vector_type(8)))  unsigned int v8u;
typedef __attribute__((ext_vector_type(4)))  unsigned int v4u;

#define T_TOKENS   16384
#define HDIM       4096
#define NEXP       64
#define KC         512                   // K-chunk staged in LDS per iteration
#define NCHUNK     (HDIM / KC)           // 8
#define STEPS      (KC / 32)             // 16 WMMA K-steps per chunk
#define NWAVES     8
#define TOK_PER_WG (NWAVES * 16)         // 128
#define NBLK       (T_TOKENS / TOK_PER_WG)
#define FRAG_DW    (KC * NEXP / 2)       // 16384 dwords (64 KB) per chunk image

// f32 pair -> packed bf16 dword (lowers to v_cvt_pk_bf16_f32 on gfx1250)
static __device__ __forceinline__ unsigned int packbf(float lo, float hi) {
  v2ff f = {lo, hi};
  v2bf r = __builtin_convertvector(f, v2bf);
  return __builtin_bit_cast(unsigned int, r);
}

// ---------------------------------------------------------------------------
// Prep: build the bf16 WMMA-B fragment image of the whole weight in d_ws.
// Identical for every WG of the main kernel -> compute once (512 KB, L2-resident).
// grid = NCHUNK x 4 tiles, 256 threads. Thread tid owns bf16 pair k={2tid,2tid+1}.
// ---------------------------------------------------------------------------
__global__ __launch_bounds__(256)
void moe_router_prep_kernel(const float* __restrict__ w,
                            unsigned int* __restrict__ wsFrag)
{
  const int tid   = threadIdx.x;
  const int chunk = blockIdx.x >> 2;
  const int t     = blockIdx.x & 3;           // expert tile (16 experts)
  const int sS    = tid >> 4;                 // step within chunk
  const int jS    = (((tid >> 3) & 1) << 2) | (tid & 3); // dword within lane fragment
  const int hiS   = ((tid >> 2) & 1) << 4;    // lane half (K bit 3: K {8..15,24..31})
  const int baseS = sS * 1024 + hiS * 8 + jS;
  const float* wc = w + 2 * tid + chunk * KC;
  unsigned int* dst = wsFrag + chunk * FRAG_DW + baseS + t * 256;
  #pragma unroll
  for (int n = 0; n < 16; ++n) {
    const float2* wp = (const float2*)(wc + (size_t)(t * 16 + n) * HDIM);
    float2 wv = *wp;
    dst[n * 8] = packbf(wv.x, wv.y);
  }
}

// ---------------------------------------------------------------------------
// Main: 128 tokens/WG, logits via bf16 WMMA, fused top-8 + renorm softmax.
// ---------------------------------------------------------------------------
__global__ __launch_bounds__(256)
void moe_router_topk_kernel(const float* __restrict__ hs,
                            const unsigned int* __restrict__ wsFrag,
                            float* __restrict__ out_val,
                            int*   __restrict__ out_idx)
{
  __shared__ unsigned int ldsFrag[FRAG_DW];   // 64 KB

  const int tid  = threadIdx.x;
  const int lane = tid & 31;
  const int wave = tid >> 5;

  // A operand (16-bit A layout, ISA 7.12.2): lane L<16 -> row M=L, K {0..7,16..23};
  // lane L>=16 -> row M=L-16, K {8..15,24..31}
  const int rowA = blockIdx.x * TOK_PER_WG + wave * 16 + (lane & 15);
  const int ko   = (lane >> 4) * 8;
  const float* rowp = hs + (size_t)rowA * HDIM + ko;

  v8f acc[4] = {};   // 4 expert tiles of 16 -> 64 experts, fp32 accum

  auto load_frags = [&](int step, v16bf* bf) {
    #pragma unroll
    for (int t = 0; t < 4; ++t) {
      const v8u* bp = (const v8u*)&ldsFrag[(((step * 4 + t) * 32 + lane) << 3)];
      bf[t] = __builtin_bit_cast(v16bf, *bp);
    }
  };
  auto mma4 = [&](v8ff r0, v8ff r1, v16bf* bf) {
    v16f f = __builtin_shufflevector(r0, r1, 0,1,2,3,4,5,6,7,8,9,10,11,12,13,14,15);
    v16bf a = __builtin_convertvector(f, v16bf);
    #pragma unroll
    for (int t = 0; t < 4; ++t) {
      acc[t] = __builtin_amdgcn_wmma_f32_16x16x32_bf16(
          /*neg_a=*/false, a, /*neg_b=*/false, bf[t],
          /*c_mod=*/(short)0, acc[t], /*reuse_a=*/false, /*reuse_b=*/false);
    }
  };

  for (int chunk = 0; chunk < NCHUNK; ++chunk) {
    __syncthreads();
    // ---- flat 64 KB copy of this chunk's fragment image: global -> LDS ----
    {
      const unsigned int* src = wsFrag + chunk * FRAG_DW + (tid << 2);
      #pragma unroll
      for (int i = 0; i < 16; ++i) {
        v4u v = *(const v4u*)(src + i * 1024);        // RT: keep L2-resident
        *(v4u*)&ldsFrag[(tid << 2) + i * 1024] = v;   // conflict-free ds_store_b128
      }
    }
    __syncthreads();

    // ---- GEMM over this K-chunk: explicit 2-deep ping-pong (no reg copies) ----
    int k0 = chunk * KC;
    // hidden: streamed once, zero reuse -> non-temporal (protect L2 weight image)
    v8ff a00 = __builtin_nontemporal_load((const v8ff*)(rowp + k0));
    v8ff a01 = __builtin_nontemporal_load((const v8ff*)(rowp + k0 + 16));
    v16bf b0[4], b1[4];
    load_frags(0, b0);
    v8ff a10, a11;
    for (int sp = 0; sp + 2 < STEPS; sp += 2) {
      load_frags(sp + 1, b1);
      a10 = __builtin_nontemporal_load((const v8ff*)(rowp + k0 + 32));
      a11 = __builtin_nontemporal_load((const v8ff*)(rowp + k0 + 48));
      mma4(a00, a01, b0);                       // consumes step sp
      load_frags(sp + 2, b0);                   // refill consumed buffers
      a00 = __builtin_nontemporal_load((const v8ff*)(rowp + k0 + 64));
      a01 = __builtin_nontemporal_load((const v8ff*)(rowp + k0 + 80));
      mma4(a10, a11, b1);                       // consumes step sp+1
      k0 += 64;
    }
    // tail: steps STEPS-2, STEPS-1 (b0/a0 already hold step STEPS-2)
    load_frags(STEPS - 1, b1);
    a10 = __builtin_nontemporal_load((const v8ff*)(rowp + k0 + 32));
    a11 = __builtin_nontemporal_load((const v8ff*)(rowp + k0 + 48));
    mma4(a00, a01, b0);
    mma4(a10, a11, b1);
  }

  // ---- logits -> LDS (reuse fragment buffer: 128 tok x 64 exp x f32 = 32 KB) ----
  __syncthreads();
  float* lg = (float*)ldsFrag;
  {
    // C layout: VGPR i, lanes 0-15 -> M=i ; lanes 16-31 -> M=8+i ; N=lane%16
    int tl = wave * 16 + (lane >> 4) * 8;
    int ec = lane & 15;
    #pragma unroll
    for (int t = 0; t < 4; ++t)
      #pragma unroll
      for (int i = 0; i < 8; ++i)
        lg[(tl + i) * 64 + t * 16 + ec] = acc[t][i];
  }
  __syncthreads();

  // ---- fused top-8 + renormalized softmax (full-softmax denom cancels) ----
  if (tid < TOK_PER_WG) {
    const float* L = lg + tid * 64;
    float tv[8]; int ti[8];
    #pragma unroll
    for (int k = 0; k < 8; ++k) { tv[k] = -3.0e38f; ti[k] = 0; }
    for (int e = 0; e < NEXP; ++e) {       // 8-slot insertion, static indexing
      float x = L[e]; int xi = e;
      #pragma unroll
      for (int k = 0; k < 8; ++k) {
        bool sw = x > tv[k];
        float nv = sw ? x     : tv[k];  int ni = sw ? xi    : ti[k];
        float ox = sw ? tv[k] : x;      int oi = sw ? ti[k] : xi;
        tv[k] = nv; ti[k] = ni; x = ox; xi = oi;
      }
    }
    float m = tv[0], s = 0.f, ev[8];
    #pragma unroll
    for (int k = 0; k < 8; ++k) { ev[k] = __expf(tv[k] - m); s += ev[k]; }
    float inv = 1.0f / s;
    int tok = blockIdx.x * TOK_PER_WG + tid;
    #pragma unroll
    for (int k = 0; k < 8; ++k) {
      out_val[tok * 8 + k] = ev[k] * inv;
      out_idx[tok * 8 + k] = ti[k];
    }
  }
}

extern "C" void kernel_launch(void* const* d_in, const int* in_sizes, int n_in,
                              void* d_out, int out_size, void* d_ws, size_t ws_size,
                              hipStream_t stream) {
  const float* hs = (const float*)d_in[0];   // [16384, 4096] f32
  const float* w  = (const float*)d_in[1];   // [64, 4096] f32
  unsigned int* wsFrag = (unsigned int*)d_ws;             // 512 KB fragment image
  float* out_val = (float*)d_out;                         // [16384, 8] f32
  int*   out_idx = (int*)d_out + (size_t)T_TOKENS * 8;    // [16384, 8] i32
  moe_router_prep_kernel<<<dim3(NCHUNK * 4), dim3(256), 0, stream>>>(w, wsFrag);
  moe_router_topk_kernel<<<dim3(NBLK), dim3(256), 0, stream>>>(hs, wsFrag, out_val, out_idx);
}
